// VQEmbedding_85177791414763
// MI455X (gfx1250) — matrix operand order repaired and checked by reading, not compiled, not run on hardware
//
#include <hip/hip_runtime.h>

// ---------------------------------------------------------------------------
// VQ embedding (straight-through) for MI455X / gfx1250.
//   scores = z_half @ W_halfT  (bf16 WMMA, f32 accum)
//   idx    = argmax_k scores   (fused: packed-key global_atomic_max_u64)
//   z_q    = W[idx]            (gather kernel)
// HBM-bound on the 512MB score write (~30us floor @ 23.3TB/s). L2 tile
// re-read traffic minimized via one-time bf16 pre-conversion (ws permitting),
// a 256x128 block tile (each B fragment feeds two A strips), and NT temporal
// hints on the write-once score/output streams so they don't evict the
// L2-resident bf16 tiles.
// ---------------------------------------------------------------------------

#define KN   8192   // codebook entries
#define DN   512    // embedding dim per half
#define BN   8192   // batch
#define BM   256    // tile rows (batch)
#define BNC  128    // tile cols (codes)
#define KT   32     // K-step (bf16 WMMA K)
#define ASTR 40     // LDS tile row stride in bf16 elems (32 + 8 pad = 80B, 16B aligned)
#define CSTR 132    // LDS stage row stride in f32 (128 + 4 pad)

typedef __attribute__((ext_vector_type(16))) __bf16 v16bf;
typedef __attribute__((ext_vector_type(8)))  __bf16 v8bf;
typedef __attribute__((ext_vector_type(8)))  float  v8f;
typedef __attribute__((ext_vector_type(4)))  float  v4f;
typedef __attribute__((ext_vector_type(4)))  unsigned short v4us;
typedef __attribute__((ext_vector_type(8)))  unsigned short v8us;

__device__ __forceinline__ unsigned short f2bf(float f) {
    unsigned u = __float_as_uint(f);
    unsigned r = u + 0x7FFFu + ((u >> 16) & 1u);   // round-to-nearest-even
    return (unsigned short)(r >> 16);
}

// order-preserving f32 -> u32 map (larger float => larger uint)
__device__ __forceinline__ unsigned ford(float f) {
    unsigned u = __float_as_uint(f);
    return (u & 0x80000000u) ? ~u : (u | 0x80000000u);
}

// A fragment (16x32 bf16): lane m(0-15) row m: v0..3=K0..7, v4..7=K16..23;
// lanes 16-31: K8..15, K24..31.  (ISA 7.12.2)
__device__ __forceinline__ v16bf load_fragA(const unsigned short* tile, int lane, int strip) {
    int row = strip + (lane & 15);
    int c0  = (lane >> 4) << 3;                    // 0 or 8
    v8bf lo = *(const v8bf*)(tile + row * ASTR + c0);
    v8bf hi = *(const v8bf*)(tile + row * ASTR + c0 + 16);
    v16bf r;
#pragma unroll
    for (int i = 0; i < 8; ++i) { r[i] = lo[i]; r[i + 8] = hi[i]; }
    return r;
}

// B fragment (32x16 bf16, col n = W row n): lane n(0-15): K0..7, K16..23;
// lanes 16-31: K8..15, K24..31 (same k order as A).
__device__ __forceinline__ v16bf load_fragB(const unsigned short* tile, int lane, int strip) {
    int row = strip + (lane & 15);
    int c0  = (lane >> 4) << 3;                    // 0 or 8
    v8bf lo = *(const v8bf*)(tile + row * ASTR + c0);
    v8bf hi = *(const v8bf*)(tile + row * ASTR + c0 + 16);
    v16bf r;
#pragma unroll
    for (int i = 0; i < 8; ++i) { r[i] = lo[i]; r[i + 8] = hi[i]; }
    return r;
}

__global__ void vq_init_kernel(unsigned long long* amax, int n) {
    int i = blockIdx.x * blockDim.x + threadIdx.x;
    if (i < n) amax[i] = 0ull;
}

// one-time f32 -> bf16 conversion (8 elements / thread, 128-bit ld / 128-bit st)
// NT loads: the f32 sources are read exactly once, keep them out of L2; the
// bf16 destinations are re-read 32-64x by the GEMM, so stores stay cached.
__global__ void vq_cvt_kernel(const float* __restrict__ src,
                              unsigned short* __restrict__ dst, int n8) {
    int i = blockIdx.x * blockDim.x + threadIdx.x;
    if (i >= n8) return;
    v4f a = __builtin_nontemporal_load((const v4f*)(src + (size_t)i * 8));
    v4f b = __builtin_nontemporal_load((const v4f*)(src + (size_t)i * 8 + 4));
    v8us p;
    p[0] = f2bf(a.x); p[1] = f2bf(a.y); p[2] = f2bf(a.z); p[3] = f2bf(a.w);
    p[4] = f2bf(b.x); p[5] = f2bf(b.y); p[6] = f2bf(b.z); p[7] = f2bf(b.w);
    *(v8us*)(dst + (size_t)i * 8) = p;
}

// PRE=true: sources are pre-converted bf16 (ushort) arrays.
// PRE=false: sources are the original f32 arrays, convert while staging.
template <bool PRE>
__global__ __launch_bounds__(256)
void vq_gemm_kernel(const void* __restrict__ zsrc,
                    const void* __restrict__ wasrc,
                    const void* __restrict__ wvsrc,
                    float* __restrict__ out,
                    unsigned long long* __restrict__ amax) {
    __shared__ unsigned short shA[BM * ASTR];    // 256x32 bf16 (+pad)
    __shared__ unsigned short shB[BNC * ASTR];   // 128x32 bf16 (+pad)
    __shared__ float          shC[128 * CSTR];   // store stage (one 128-row pass)

    const int t    = threadIdx.x;
    const int wave = t >> 5;
    const int lane = t & 31;
    const int half = blockIdx.z;
    const int rowM0 = blockIdx.y * BM;     // batch-row base
    const int colN0 = blockIdx.x * BNC;    // code-col base

    v8f acc[2][8];
#pragma unroll
    for (int h = 0; h < 2; ++h)
#pragma unroll
        for (int n = 0; n < 8; ++n) acc[h][n] = (v8f)0.f;

    for (int d0 = 0; d0 < DN; d0 += KT) {
        if (PRE) {
            const unsigned short* zg = (const unsigned short*)zsrc;
            const unsigned short* wg =
                (const unsigned short*)(half ? wvsrc : wasrc);
            const int row = t >> 2;          // 0..63
            const int cc  = (t & 3) * 8;     // 16B chunk within 32-elem row
#pragma unroll
            for (int p = 0; p < 4; ++p) {    // A: 256 rows
                int r = row + p * 64;
                v8us v = *(const v8us*)(zg + (size_t)(rowM0 + r) * (2 * DN) +
                                        (size_t)half * DN + d0 + cc);
                *(v8us*)&shA[r * ASTR + cc] = v;
            }
#pragma unroll
            for (int p = 0; p < 2; ++p) {    // B: 128 rows
                int r = row + p * 64;
                v8us v = *(const v8us*)(wg + (size_t)(colN0 + r) * DN + d0 + cc);
                *(v8us*)&shB[r * ASTR + cc] = v;
            }
        } else {
            const float* zg = (const float*)zsrc + (size_t)half * DN;
            const float* wg = (const float*)(half ? wvsrc : wasrc);
            const int row = t >> 1;          // 0..127
            const int lc  = (t & 1) * 16;
#pragma unroll
            for (int p = 0; p < 2; ++p) {    // A: 256 rows
                int r = row + p * 128;
                const float* g = zg + (size_t)(rowM0 + r) * (2 * DN) + d0 + lc;
#pragma unroll
                for (int i = 0; i < 4; ++i) {
                    v4f v = *(const v4f*)(g + i * 4);
                    v4us q;
                    q.x = f2bf(v.x); q.y = f2bf(v.y); q.z = f2bf(v.z); q.w = f2bf(v.w);
                    *(v4us*)&shA[r * ASTR + lc + i * 4] = q;
                }
            }
            {                                 // B: 128 rows
                const float* g = wg + (size_t)(colN0 + row) * DN + d0 + lc;
#pragma unroll
                for (int i = 0; i < 4; ++i) {
                    v4f v = *(const v4f*)(g + i * 4);
                    v4us q;
                    q.x = f2bf(v.x); q.y = f2bf(v.y); q.z = f2bf(v.z); q.w = f2bf(v.w);
                    *(v4us*)&shB[row * ASTR + lc + i * 4] = q;
                }
            }
        }
        __syncthreads();

        v16bf afr0 = load_fragA(shA, lane, wave * 32);
        v16bf afr1 = load_fragA(shA, lane, wave * 32 + 16);
#pragma unroll
        for (int n = 0; n < 8; ++n) {
            v16bf bfr = load_fragB(shB, lane, n * 16);
            acc[0][n] = __builtin_amdgcn_wmma_f32_16x16x32_bf16(
                false, afr0, false, bfr, (short)0, acc[0][n], false, false);
            acc[1][n] = __builtin_amdgcn_wmma_f32_16x16x32_bf16(
                false, afr1, false, bfr, (short)0, acc[1][n], false, false);
        }
        __syncthreads();
    }

    const int lgrp = lane >> 4;   // 0/1
    const int lcol = lane & 15;

    // ---- fused per-row argmax: packed (ord(score)<<32 | col) atomic max ----
    unsigned long long* am = amax + (size_t)half * BN;
#pragma unroll
    for (int h = 0; h < 2; ++h) {
#pragma unroll
        for (int r = 0; r < 8; ++r) {
            float best = acc[h][0][r];
            int   bn   = 0;
#pragma unroll
            for (int n = 1; n < 8; ++n) {
                float v = acc[h][n][r];
                if (v > best) { best = v; bn = n; }
            }
            int bcol = colN0 + bn * 16 + lcol;
#pragma unroll
            for (int m = 1; m < 16; m <<= 1) {
                float ov = __shfl_xor(best, m, 32);
                int   oc = __shfl_xor(bcol, m, 32);
                if (ov > best || (ov == best && oc > bcol)) { best = ov; bcol = oc; }
            }
            if (lcol == 0) {
                int gm = rowM0 + wave * 32 + h * 16 + lgrp * 8 + r;
                unsigned long long key =
                    ((unsigned long long)ford(best) << 32) | (unsigned)bcol;
                atomicMax(&am[gm], key);
            }
        }
    }

    // ---- stage scores through LDS for coalesced 128-bit NT stores ----------
    // Two passes: pass p stages every wave's h==p strip (slot row = wave*16+j,
    // global row = rowM0 + wave*32 + p*16 + j). Scores are write-once /
    // read-never (argmax already fused) -> non-temporal so the 512MB stream
    // doesn't evict the L2-resident bf16 tiles.
    float* outS = out + (size_t)2 * BN * (2 * DN);  // scores region
#pragma unroll
    for (int p = 0; p < 2; ++p) {
        __syncthreads();
#pragma unroll
        for (int n = 0; n < 8; ++n)
#pragma unroll
            for (int r = 0; r < 8; ++r)
                shC[(wave * 16 + lgrp * 8 + r) * CSTR + n * 16 + lcol] =
                    acc[p][n][r];
        __syncthreads();
#pragma unroll
        for (int i = 0; i < 16; ++i) {
            int flat = t + i * 256;          // float4 id within 128x128
            int srow = flat >> 5;
            int c4   = (flat & 31) * 4;
            int grow = rowM0 + (srow >> 4) * 32 + p * 16 + (srow & 15);
            v4f v = *(const v4f*)&shC[srow * CSTR + c4];
            __builtin_nontemporal_store(
                v, (v4f*)(outS + (size_t)grow * (2 * KN) +
                          (size_t)half * KN + colN0 + c4));
        }
    }
}

__global__ __launch_bounds__(128)
void vq_gather_kernel(const float* __restrict__ Wa,
                      const float* __restrict__ Wv,
                      const unsigned long long* __restrict__ amax,
                      float* __restrict__ out) {
    const int b    = blockIdx.x;
    const int half = blockIdx.y;
    const int t    = threadIdx.x;  // 128 threads * float4 = 512 floats
    unsigned idx = (unsigned)(amax[(size_t)half * BN + b] & 0xFFFFFFFFull);
    const float* Wr = (half ? Wv : Wa) + (size_t)idx * DN;
    v4f v = *(const v4f*)(Wr + t * 4);
    size_t o = (size_t)b * (2 * DN) + (size_t)half * DN + t * 4;
    __builtin_nontemporal_store(v, (v4f*)(out + o));                         // z_q_x_st
    __builtin_nontemporal_store(v, (v4f*)(out + (size_t)BN * (2 * DN) + o)); // z_q_x
}

extern "C" void kernel_launch(void* const* d_in, const int* in_sizes, int n_in,
                              void* d_out, int out_size, void* d_ws, size_t ws_size,
                              hipStream_t stream) {
    const float* z  = (const float*)d_in[0];   // [B, 2D]
    const float* Wa = (const float*)d_in[1];   // [K, D]
    const float* Wv = (const float*)d_in[2];   // [K, D]
    float* out = (float*)d_out;

    unsigned long long* amax = (unsigned long long*)d_ws;  // 2*B u64 = 128KB
    const size_t amaxBytes = (size_t)2 * BN * sizeof(unsigned long long);
    const size_t zElems = (size_t)BN * 2 * DN;   // 8.4M
    const size_t wElems = (size_t)KN * DN;       // 4.2M
    const size_t need   = amaxBytes + (zElems + 2 * wElems) * sizeof(unsigned short);

    // zero argmax keys (ws is poisoned; must re-init deterministically)
    vq_init_kernel<<<(2 * BN + 255) / 256, 256, 0, stream>>>(amax, 2 * BN);

    dim3 gg(KN / BNC, BN / BM, 2);   // (64, 32, 2)

    if (ws_size >= need) {
        // one-time bf16 pre-conversion: halves all L2 tile re-read traffic
        unsigned short* zbf  = (unsigned short*)((char*)d_ws + amaxBytes);
        unsigned short* wabf = zbf + zElems;
        unsigned short* wvbf = wabf + wElems;
        int zn8 = (int)(zElems / 8), wn8 = (int)(wElems / 8);
        vq_cvt_kernel<<<(zn8 + 255) / 256, 256, 0, stream>>>(z, zbf, zn8);
        vq_cvt_kernel<<<(wn8 + 255) / 256, 256, 0, stream>>>(Wa, wabf, wn8);
        vq_cvt_kernel<<<(wn8 + 255) / 256, 256, 0, stream>>>(Wv, wvbf, wn8);
        vq_gemm_kernel<true><<<gg, 256, 0, stream>>>(zbf, wabf, wvbf, out, amax);
    } else {
        vq_gemm_kernel<false><<<gg, 256, 0, stream>>>(z, Wa, Wv, out, amax);
    }

    dim3 gh(BN, 2, 1);
    vq_gather_kernel<<<gh, 128, 0, stream>>>(Wa, Wv, amax, out);
}
